// QuantModuleINT4_9706626089387
// MI455X (gfx1250) — 8-bit matrix operand order repaired and checked by reading, compiled, not run on hardware
//
#include <hip/hip_runtime.h>
#include <hip/hip_bf16.h>

typedef __attribute__((ext_vector_type(8))) int v8i;
typedef __attribute__((ext_vector_type(4))) int v4i;
typedef __attribute__((ext_vector_type(2))) int v2i;

#define QEPS 1e-8f

__device__ __forceinline__ unsigned enc_f(float f) {
    unsigned u = __float_as_uint(f);
    return (u & 0x80000000u) ? ~u : (u | 0x80000000u);
}
__device__ __forceinline__ float dec_f(unsigned e) {
    unsigned u = (e & 0x80000000u) ? (e & 0x7FFFFFFFu) : ~e;
    return __uint_as_float(u);
}

__global__ void init_mm_kernel(unsigned* mm) {
    mm[0] = 0xFFFFFFFFu;
    mm[1] = 0x00000000u;
}

__global__ void act_minmax_kernel(const float* __restrict__ x, unsigned* mm, size_t n) {
    size_t stride = (size_t)gridDim.x * blockDim.x;
    size_t i = (size_t)blockIdx.x * blockDim.x + threadIdx.x;
    float mn = 3.4e38f, mx = -3.4e38f;
    for (; i < n; i += stride) {
        float v = x[i];
        mn = fminf(mn, v);
        mx = fmaxf(mx, v);
    }
    for (int off = 16; off > 0; off >>= 1) {
        mn = fminf(mn, __shfl_xor(mn, off, 32));
        mx = fmaxf(mx, __shfl_xor(mx, off, 32));
    }
    if ((threadIdx.x & 31) == 0) {
        atomicMin(&mm[0], enc_f(mn));
        atomicMax(&mm[1], enc_f(mx));
    }
}

__global__ void __launch_bounds__(256) wquant_kernel(const float* __restrict__ w,
                                                     signed char* __restrict__ wq,
                                                     float* __restrict__ sW, int K) {
    int row = blockIdx.x;
    const float* wr = w + (size_t)row * K;
    float mn = 3.4e38f, mx = -3.4e38f;
    for (int i = threadIdx.x; i < K; i += 256) {
        float v = wr[i];
        mn = fminf(mn, v);
        mx = fmaxf(mx, v);
    }
    for (int off = 16; off > 0; off >>= 1) {
        mn = fminf(mn, __shfl_xor(mn, off, 32));
        mx = fmaxf(mx, __shfl_xor(mx, off, 32));
    }
    __shared__ float smn[8], smx[8];
    int wave = threadIdx.x >> 5;
    if ((threadIdx.x & 31) == 0) { smn[wave] = mn; smx[wave] = mx; }
    __syncthreads();
    mn = smn[0]; mx = smx[0];
    #pragma unroll
    for (int j = 1; j < 8; ++j) { mn = fminf(mn, smn[j]); mx = fmaxf(mx, smx[j]); }

    float scale = (mx - mn) * (1.0f / 15.0f);
    if (scale < QEPS) scale = QEPS;
    float zp = rintf(-mn / scale);
    if (threadIdx.x == 0) sW[row] = scale;

    float inv_s = 1.0f / scale;
    signed char* wqr = wq + (size_t)row * K;
    for (int i = threadIdx.x; i < K; i += 256) {
        float q = rintf(wr[i] * inv_s) + zp;
        q = fminf(fmaxf(q, 0.0f), 15.0f);
        wqr[i] = (signed char)(q - zp);
    }
}

__global__ void act_quant_kernel(const float* __restrict__ x, signed char* __restrict__ xq,
                                 const unsigned* __restrict__ mm, size_t n) {
    float mn = dec_f(mm[0]), mx = dec_f(mm[1]);
    float scale = fmaxf((mx - mn) * (1.0f / 15.0f), QEPS);
    float zp = rintf(-mn / scale);
    float inv_s = 1.0f / scale;
    size_t stride = (size_t)gridDim.x * blockDim.x;
    for (size_t i = (size_t)blockIdx.x * blockDim.x + threadIdx.x; i < n; i += stride) {
        float q = rintf(x[i] * inv_s) + zp;
        q = fminf(fmaxf(q, 0.0f), 15.0f);
        xq[i] = (signed char)(q - zp);
    }
}

#define LDS_STRIDE 80
#define LDS_A_BYTES (128 * LDS_STRIDE)
#define LDS_B_BYTES (64 * LDS_STRIDE)
#define LDS_BUF_BYTES (LDS_A_BYTES + LDS_B_BYTES)

__global__ void __launch_bounds__(256) gemm_i4_wmma_kernel(
    const signed char* __restrict__ xq, const signed char* __restrict__ wq,
    const float* __restrict__ sW, const unsigned* __restrict__ mm,
    const float* __restrict__ bias, float* __restrict__ y,
    int M, int N, int K) {
    __shared__ __align__(16) signed char lds[2][LDS_BUF_BYTES];

    const int tid  = threadIdx.x;
    const int lane = tid & 31;
    const int wave = tid >> 5;
    const int half = lane >> 4;
    const int l16  = lane & 15;

    const int mBlock = blockIdx.y * 128;
    const int nBase  = blockIdx.x * 64;

    const int cRow = tid >> 2;
    const int cOff = (tid & 3) << 4;
    const size_t gA0 = (size_t)(mBlock + cRow) * K + cOff;
    const size_t gA1 = (size_t)(mBlock + cRow + 64) * K + cOff;
    const size_t gB  = (size_t)(nBase + cRow) * K + cOff;
    const int sA0 = cRow * LDS_STRIDE + cOff;
    const int sA1 = sA0 + 64 * LDS_STRIDE;
    const int sB  = cRow * LDS_STRIDE + cOff;

    const int aOff = (wave * 16 + l16) * LDS_STRIDE + half * 16;
    const int bOff = l16 * LDS_STRIDE + half * 16;

    v8i zero = {0, 0, 0, 0, 0, 0, 0, 0};
    v8i acc[4];
    #pragma unroll
    for (int t = 0; t < 4; ++t) acc[t] = zero;

    {
        signed char* bufA = &lds[0][0];
        signed char* bufB = &lds[0][LDS_A_BYTES];
        *(v4i*)(bufA + sA0) = *(const v4i*)(xq + gA0);
        *(v4i*)(bufA + sA1) = *(const v4i*)(xq + gA1);
        *(v4i*)(bufB + sB)  = *(const v4i*)(wq + gB);
    }
    __syncthreads();

    const int ksteps = K >> 6;
    for (int ks = 0; ks < ksteps; ++ks) {
        const signed char* cA = &lds[ks & 1][0];
        const signed char* cB = &lds[ks & 1][LDS_A_BYTES];

        v4i pA0, pA1, pB;
        const bool has_next = (ks + 1) < ksteps;
        if (has_next) {
            const int k1 = (ks + 1) << 6;
            pA0 = *(const v4i*)(xq + gA0 + k1);
            pA1 = *(const v4i*)(xq + gA1 + k1);
            pB  = *(const v4i*)(wq + gB + k1);
        }

        v8i a;
        {
            const signed char* p = cA + aOff;
            v2i d0 = *(const v2i*)(p);
            v2i d1 = *(const v2i*)(p + 8);
            v2i d2 = *(const v2i*)(p + 32);
            v2i d3 = *(const v2i*)(p + 40);
            a[0] = d0[0]; a[1] = d0[1];
            a[2] = d1[0]; a[3] = d1[1];
            a[4] = d2[0]; a[5] = d2[1];
            a[6] = d3[0]; a[7] = d3[1];
        }

        v8i bfr[4];
        #pragma unroll
        for (int t = 0; t < 4; ++t) {
            const signed char* p = cB + t * (16 * LDS_STRIDE) + bOff;
            v4i lo = *(const v4i*)(p);
            v4i hi = *(const v4i*)(p + 32);
            bfr[t][0] = lo[0]; bfr[t][1] = lo[1]; bfr[t][2] = lo[2]; bfr[t][3] = lo[3];
            bfr[t][4] = hi[0]; bfr[t][5] = hi[1]; bfr[t][6] = hi[2]; bfr[t][7] = hi[3];
        }
        #pragma unroll
        for (int t = 0; t < 4; ++t) {
            acc[t] = __builtin_amdgcn_wmma_i32_16x16x64_iu8(
                true, a, true, bfr[t], acc[t], false, false);
        }

        if (has_next) {
            signed char* nA = &lds[(ks + 1) & 1][0];
            signed char* nB = &lds[(ks + 1) & 1][LDS_A_BYTES];
            *(v4i*)(nA + sA0) = pA0;
            *(v4i*)(nA + sA1) = pA1;
            *(v4i*)(nB + sB)  = pB;
        }
        __syncthreads();
    }

    float mn = dec_f(mm[0]), mx = dec_f(mm[1]);
    float sX = fmaxf((mx - mn) * (1.0f / 15.0f), QEPS);
    const int mBase = mBlock + wave * 16;
    #pragma unroll
    for (int t = 0; t < 4; ++t) {
        int col = nBase + t * 16 + l16;
        float s = sX * sW[col];
        float bi = bias[col];
        #pragma unroll
        for (int g = 0; g < 8; ++g) {
            int row = mBase + half * 8 + g;
            y[(size_t)row * N + col] = s * (float)acc[t][g] + bi;
        }
    }
}

extern "C" void kernel_launch(void* const* d_in, const int* in_sizes, int n_in,
                              void* d_out, int out_size, void* d_ws, size_t ws_size,
                              hipStream_t stream) {
    const float* x    = (const float*)d_in[0];
    const float* w    = (const float*)d_in[1];
    const float* bias = (const float*)d_in[2];
    float* y = (float*)d_out;

    const int N = in_sizes[2];
    const int K = in_sizes[1] / N;
    const int M = in_sizes[0] / K;

    unsigned* mm = (unsigned*)d_ws;
    float* sW = (float*)((char*)d_ws + 256);
    signed char* wq = (signed char*)((char*)d_ws + 256 + (size_t)N * sizeof(float));
    signed char* xq = wq + (size_t)N * K;

    const size_t nx = (size_t)M * K;

    init_mm_kernel<<<1, 1, 0, stream>>>(mm);
    act_minmax_kernel<<<2048, 256, 0, stream>>>(x, mm, nx);
    wquant_kernel<<<N, 256, 0, stream>>>(w, wq, sW, K);
    act_quant_kernel<<<2048, 256, 0, stream>>>(x, xq, mm, nx);

    dim3 grid(N / 64, M / 128);
    gemm_i4_wmma_kernel<<<grid, 256, 0, stream>>>(xq, wq, sW, mm, bias, y, M, N, K);
}
